// LinearRowShared4Bit_7078106104436
// MI455X (gfx1250) — matrix operand order repaired and checked by reading, not compiled, not run on hardware
//
#include <hip/hip_runtime.h>

typedef _Float16 v16h __attribute__((ext_vector_type(16)));
typedef _Float16 h2   __attribute__((ext_vector_type(2)));
typedef float    v8f  __attribute__((ext_vector_type(8)));
typedef unsigned int u32;
typedef u32 u32x4 __attribute__((ext_vector_type(4)));

#define M_DIM   16
#define O_DIM   8192
#define I_DIM   8192
#define KBLOCKS 256          // I_DIM / 32 (dequant groups along K)
#define ROW_U32 4096         // int32 words per weight row (256 blocks * 16)
#define WPB     4            // waves per block (wave32)
#define TILES   (O_DIM / 16) // 512 output tiles of 16 features

// ---------------------------------------------------------------------------
// dequant: packed byte b -> two f16 weights  w = (2*norm/15)*q - norm
// magic: 0x6400 | q is the f16 encoding of (1024 + q) exactly, q in [0,15]
// ---------------------------------------------------------------------------
__device__ __forceinline__ h2 deq4(u32 b, h2 s2, h2 c2) {
    u32 t = (b & 0xFu) | ((b & 0xF0u) << 12) | 0x64006400u;
    h2 q = __builtin_bit_cast(h2, t);
    const h2 k1024 = { (_Float16)1024.0f, (_Float16)1024.0f };
    h2 qa = q - k1024;          // exact: recovers q in [0,15]
    return qa * s2 + c2;        // v_pk_fma_f16: (2n/15)*q - n
}

// ---------------------------------------------------------------------------
// x fp32 -> fp16 staging (one pass, 256 KB, L2-resident afterwards)
// ---------------------------------------------------------------------------
__global__ void cvt_x_kernel(const float* __restrict__ x,
                             _Float16* __restrict__ xh, int n4) {
    int i = blockIdx.x * blockDim.x + threadIdx.x;
    if (i < n4) {
        float4 v = reinterpret_cast<const float4*>(x)[i];
        h2 lo = { (_Float16)v.x, (_Float16)v.y };
        h2 hi = { (_Float16)v.z, (_Float16)v.w };
        uint2 st;
        st.x = __builtin_bit_cast(u32, lo);
        st.y = __builtin_bit_cast(u32, hi);
        reinterpret_cast<uint2*>(xh)[i] = st;
    }
}

// ---------------------------------------------------------------------------
// Main WMMA kernel: one wave = one 16-feature output tile, K-chunk on grid.y.
// A (16x32 f16) = x tile, B (32x16 f16) = dequantized W^T tile.
//   A lane layout: lanes 0-15 row m=lane, K 0..15; lanes 16-31 K 16..31
//   B lane layout: lanes 0-15 col n=lane, K 0..15; lanes 16-31 K 16..31
// ---------------------------------------------------------------------------
__global__ void __launch_bounds__(WPB * 32)
wmma_q4_kernel(const u32* __restrict__ wq,
               const unsigned short* __restrict__ wn,
               const _Float16* __restrict__ xh,
               float* __restrict__ dst,
               const float* __restrict__ bias,   // null => partial mode
               int kb_per_chunk) {
    const int lane    = threadIdx.x & 31;
    const int wave    = threadIdx.x >> 5;
    const int tile    = blockIdx.x * WPB + wave;
    const int obase   = tile << 4;
    const int n       = lane & 15;
    const int halfsel = lane >> 4;          // 0: K 0..15 window, 1: K 16..31
    const int chunk   = blockIdx.y;
    const int kb0     = chunk * kb_per_chunk;

    // per-kb stride in u32x4 units is 4 (64 bytes = one 32-weight group row-slice)
    const u32x4* wrow = (const u32x4*)(wq + (size_t)(obase + n) * ROW_U32 + halfsel * 8)
                        + (size_t)kb0 * 4;
    const u32x4* xrow = (const u32x4*)(xh + (size_t)n * I_DIM + halfsel * 16)
                        + (size_t)kb0 * 4;
    const u32x4* nrow = (const u32x4*)(wn + (size_t)(obase >> 4) * KBLOCKS + kb0);

    v8f acc = {};
    const int nsb = kb_per_chunk >> 3;      // super-blocks of 8 k-blocks
    for (int sb = 0; sb < nsb; ++sb) {
        __builtin_prefetch(wrow + (size_t)(sb + 2) * 32, 0, 0);  // global_prefetch_b8
        u32x4 nr = nrow[sb];                // 8 f16 norms for this super-block
#pragma unroll
        for (int j = 0; j < 8; ++j) {
            const int kb = sb * 8 + j;
            u32 nb = (j & 1) ? (nr[j >> 1] >> 16) : (nr[j >> 1] & 0xffffu);
            _Float16 nh = __builtin_bit_cast(_Float16, (unsigned short)nb);
            _Float16 sh = nh * (_Float16)(2.0f / 15.0f);
            h2 s2 = { sh, sh };
            h2 c2 = { -nh, -nh };           // -7.5 * (2n/15) == -n exactly

            u32x4 w0 = wrow[(size_t)kb * 4];       // 4 packed bytes (one per int32) = 8 weights
            u32x4 w1 = wrow[(size_t)kb * 4 + 1];

            union { v16h v; h2 p[8]; } B;
            B.p[0] = deq4(w0[0], s2, c2);
            B.p[1] = deq4(w0[1], s2, c2);
            B.p[2] = deq4(w0[2], s2, c2);
            B.p[3] = deq4(w0[3], s2, c2);
            B.p[4] = deq4(w1[0], s2, c2);
            B.p[5] = deq4(w1[1], s2, c2);
            B.p[6] = deq4(w1[2], s2, c2);
            B.p[7] = deq4(w1[3], s2, c2);

            union { v16h v; u32x4 q[2]; } A;
            A.q[0] = xrow[(size_t)kb * 4];         // K {0..7} of this lane's window
            A.q[1] = xrow[(size_t)kb * 4 + 1];     // K {8..15} of this lane's window

            acc = __builtin_amdgcn_wmma_f32_16x16x32_f16(
                false, A.v, false, B.v, (short)0, acc, false, false);
        }
    }

    // C/D layout: VGPR r -> row m = r + 8*halfsel, col = lane&15
    float bv = 0.0f;
    if (bias) bv = bias[obase + n];
    float* drow = dst + (size_t)chunk * (M_DIM * O_DIM)
                      + (size_t)(halfsel * 8) * O_DIM + obase + n;
#pragma unroll
    for (int r = 0; r < 8; ++r)
        drow[(size_t)r * O_DIM] = acc[r] + bv;
}

// ---------------------------------------------------------------------------
// Deterministic partial reduction + bias
// ---------------------------------------------------------------------------
__global__ void reduce_bias_kernel(const float* __restrict__ part,
                                   const float* __restrict__ bias,
                                   float* __restrict__ out, int nchunk) {
    int i = blockIdx.x * blockDim.x + threadIdx.x;  // < M_DIM*O_DIM
    float s = bias[i & (O_DIM - 1)];
    for (int c = 0; c < nchunk; ++c)
        s += part[(size_t)c * (M_DIM * O_DIM) + i];
    out[i] = s;
}

extern "C" void kernel_launch(void* const* d_in, const int* in_sizes, int n_in,
                              void* d_out, int out_size, void* d_ws, size_t ws_size,
                              hipStream_t stream) {
    const float*          x    = (const float*)d_in[0];
    const u32*            wq   = (const u32*)d_in[1];            // packed bytes as int32
    const unsigned short* wn   = (const unsigned short*)d_in[2]; // fp16 bits
    const float*          bias = (const float*)d_in[3];
    float*                out  = (float*)d_out;

    _Float16* xh = (_Float16*)d_ws;
    const size_t X_BYTES    = (size_t)M_DIM * I_DIM * sizeof(_Float16); // 256 KB
    const size_t PART_BYTES = (size_t)M_DIM * O_DIM * sizeof(float);    // 512 KB
    float* part = (float*)((char*)d_ws + X_BYTES);

    cvt_x_kernel<<<(M_DIM * I_DIM / 4 + 255) / 256, 256, 0, stream>>>(
        x, xh, M_DIM * I_DIM / 4);

    const int KSPLIT = 4;
    if (ws_size >= X_BYTES + (size_t)KSPLIT * PART_BYTES) {
        // split K for 4x wave concurrency; deterministic two-phase reduction
        dim3 grid(TILES / WPB, KSPLIT);
        wmma_q4_kernel<<<grid, WPB * 32, 0, stream>>>(
            wq, wn, xh, part, nullptr, KBLOCKS / KSPLIT);
        reduce_bias_kernel<<<(M_DIM * O_DIM) / 256, 256, 0, stream>>>(
            part, bias, out, KSPLIT);
    } else {
        // fallback: single pass straight into out with bias
        dim3 grid(TILES / WPB, 1);
        wmma_q4_kernel<<<grid, WPB * 32, 0, stream>>>(
            wq, wn, xh, out, bias, KBLOCKS);
    }
}
